// GATv2Model_19585050870259
// MI455X (gfx1250) — matrix operand order repaired and checked by reading, not compiled, not run on hardware
//
#include <hip/hip_runtime.h>
#include <hip/hip_fp16.h>
#include <math.h>

typedef _Float16 half_t;
typedef __attribute__((ext_vector_type(16))) _Float16 v16h;
typedef __attribute__((ext_vector_type(8)))  _Float16 v8h;
typedef __attribute__((ext_vector_type(8)))  float    v8f;

#define NEG_SLOPE 0.2f

// ---------------- utility kernels ----------------

__global__ void k_fill_f32(float* __restrict__ p, float v, long n) {
  long i = (long)blockIdx.x * blockDim.x + threadIdx.x;
  if (i < n) p[i] = v;
}

__global__ void k_f32_to_f16(const float* __restrict__ in, half_t* __restrict__ out, long n) {
  long i = (long)blockIdx.x * blockDim.x + threadIdx.x;
  if (i < n) out[i] = (half_t)in[i];
}

__global__ void k_pad_b_f32(const float* __restrict__ b, float* __restrict__ out,
                            int Nsrc, int Npad) {
  int n = threadIdx.x;
  if (n < Npad) out[n] = (n < Nsrc) ? b[n] : 0.0f;
}

// Pack f32 weight W[K, Nsrc] into WMMA-fragment-major f16 buffer (cols >= Nsrc zeroed).
// Layout: Bp[tile][lane][i], tile = kt*tilesN + tn (512 halves per tile), where each
// lane's element i corresponds to B-matrix entry K = kt*32 + (lane>>4)*16 + i,
// N = tn*16 + (lane&15) -- the same k order the GEMM's A fragment uses.
__global__ void k_pack_w(const float* __restrict__ W, half_t* __restrict__ Bp,
                         int K, int Nsrc, int Nc) {
  int t = blockIdx.x * blockDim.x + threadIdx.x;
  int tilesN = Nc >> 4;
  int total = (K >> 5) * tilesN * 512;
  if (t >= total) return;
  int i    = t & 15;
  int lane = (t >> 4) & 31;
  int tile = t >> 9;
  int kt = tile / tilesN, tn = tile - kt * tilesN;
  int k = (kt << 5) + ((lane >> 4) << 4) + i;
  int n = (tn << 4) + (lane & 15);
  Bp[t] = (n < Nsrc) ? (half_t)W[(long)k * Nsrc + n] : (half_t)0.0f;
}

// ---------------- WMMA GEMM: C[M,Nc] = A[M,K](f16) @ Bpack + bias ----------------
// Packed B staged to LDS once per block via CDNA5 async global->LDS copies, then one
// wave32 computes a 16 x (TILES_N*16) output strip; A fragment reused TILES_N times.
template <int TILES_N, int KT>
__global__ void k_wmma_gemm(const half_t* __restrict__ A, const half_t* __restrict__ Bp,
                            const float* __restrict__ bias, float* __restrict__ C,
                            int M, int Nc) {
  constexpr int B_HALVES = KT * TILES_N * 512;
  constexpr int B_BYTES  = B_HALVES * 2;
  __shared__ half_t sB[B_HALVES];

  // ---- cooperative async stage of packed B into LDS (all waves participate) ----
  {
    unsigned lds_base = (unsigned)(unsigned long long)&sB[0];
    const char* gbase = (const char*)Bp;
    for (int ofs = threadIdx.x * 16; ofs < B_BYTES; ofs += blockDim.x * 16) {
      unsigned lds_addr = lds_base + (unsigned)ofs;
      const char* gaddr = gbase + ofs;
      asm volatile("global_load_async_to_lds_b128 %0, %1, off"
                   :: "v"(lds_addr), "v"(gaddr) : "memory");
    }
    asm volatile("s_wait_asynccnt 0x0" ::: "memory");
  }
  __syncthreads();

  int wave = blockIdx.x * (blockDim.x >> 5) + (threadIdx.x >> 5);
  if (wave >= (M >> 4)) return;                 // after barrier: no deadlock; EXEC all-1s
  int lane = threadIdx.x & 31;
  int hsel = lane >> 4;                          // lane-half select
  int l16  = lane & 15;
  int arow = (wave << 4) + l16;

  v8f acc[TILES_N];
#pragma unroll
  for (int tn = 0; tn < TILES_N; ++tn) acc[tn] = (v8f){};

#pragma unroll
  for (int kt = 0; kt < KT; ++kt) {
    // A 16x32 f16 fragment: two aligned 16B chunks per lane, k = 16*hsel + i (matches packed B)
    const half_t* ap = A + (long)arow * (KT * 32) + kt * 32 + (hsel << 4);
    v8h lo = *(const v8h*)ap;
    v8h hi = *(const v8h*)(ap + 8);
    v16h a = __builtin_shufflevector(lo, hi, 0, 1, 2, 3, 4, 5, 6, 7,
                                     8, 9, 10, 11, 12, 13, 14, 15);
#pragma unroll
    for (int tn = 0; tn < TILES_N; ++tn) {
      // pre-packed B fragment from LDS: one contiguous 32B chunk per lane
      v16h b = *(const v16h*)(sB + ((kt * TILES_N + tn) << 9) + (lane << 4));
      acc[tn] = __builtin_amdgcn_wmma_f32_16x16x32_f16(
          false, a, false, b, (short)0, acc[tn], false, false);
    }
  }

#pragma unroll
  for (int tn = 0; tn < TILES_N; ++tn) {
    int col = (tn << 4) + l16;
    float bv = bias[col];
#pragma unroll
    for (int r = 0; r < 8; ++r) {               // C/D: VGPR r -> row r (+8 upper lane-half)
      int row = (wave << 4) + r + (hsel << 3);
      C[(long)row * Nc + col] = acc[tn][r] + bv;
    }
  }
}

// ---------------- edge-phase kernels ----------------

__device__ inline void atomicMaxFloat(float* addr, float v) {
  int iv = __float_as_int(v);
  if (iv >= 0) atomicMax((int*)addr, iv);
  else         atomicMin((unsigned int*)addr, (unsigned int)iv);
}

// layer-1 logits, C multiple of 4, all pointers 16B aligned: float4 gathers
__global__ void k_edge_logits_v4(const float* __restrict__ xl, const float* __restrict__ xr,
                                 const float* __restrict__ eattr,
                                 const float* __restrict__ We, const float* __restrict__ be,
                                 const float* __restrict__ att,
                                 const int* __restrict__ src, const int* __restrict__ dst,
                                 float* __restrict__ logits, float* __restrict__ mx,
                                 int E, int heads, int C, int nodeStride) {
  long t = (long)blockIdx.x * blockDim.x + threadIdx.x;
  if (t >= (long)E * heads) return;
  int e = (int)(t / heads), h = (int)(t - (long)e * heads);
  int s = src[e], d = dst[e];
  float ea = eattr[e];
  const float4* pl = (const float4*)(xl + (long)s * nodeStride + h * C);
  const float4* pr = (const float4*)(xr + (long)d * nodeStride + h * C);
  const float4* pw = (const float4*)(We + h * C);
  const float4* pb = (const float4*)(be + h * C);
  const float4* pa = (const float4*)(att + h * C);
  float acc = 0.0f;
  int c4 = C >> 2;
  for (int c = 0; c < c4; ++c) {
    float4 l = pl[c], r = pr[c], w = pw[c], b = pb[c], a = pa[c];
    float v0 = l.x + r.x + ea * w.x + b.x; v0 = v0 > 0.f ? v0 : NEG_SLOPE * v0;
    float v1 = l.y + r.y + ea * w.y + b.y; v1 = v1 > 0.f ? v1 : NEG_SLOPE * v1;
    float v2 = l.z + r.z + ea * w.z + b.z; v2 = v2 > 0.f ? v2 : NEG_SLOPE * v2;
    float v3 = l.w + r.w + ea * w.w + b.w; v3 = v3 > 0.f ? v3 : NEG_SLOPE * v3;
    acc += v0 * a.x + v1 * a.y + v2 * a.z + v3 * a.w;
  }
  logits[t] = acc;
  atomicMaxFloat(&mx[(long)d * heads + h], acc);
}

// generic (layer-2, C=5)
__global__ void k_edge_logits(const float* __restrict__ xl, const float* __restrict__ xr,
                              const float* __restrict__ eattr,
                              const float* __restrict__ We, const float* __restrict__ be,
                              const float* __restrict__ att,
                              const int* __restrict__ src, const int* __restrict__ dst,
                              float* __restrict__ logits, float* __restrict__ mx,
                              int E, int heads, int C, int nodeStride) {
  long t = (long)blockIdx.x * blockDim.x + threadIdx.x;
  if (t >= (long)E * heads) return;
  int e = (int)(t / heads), h = (int)(t - (long)e * heads);
  int s = src[e], d = dst[e];
  float ea = eattr[e];
  const float* pl = xl + (long)s * nodeStride + h * C;
  const float* pr = xr + (long)d * nodeStride + h * C;
  float acc = 0.0f;
  for (int c = 0; c < C; ++c) {
    float v = pl[c] + pr[c] + ea * We[h * C + c] + be[h * C + c];
    v = v > 0.0f ? v : NEG_SLOPE * v;
    acc += v * att[h * C + c];
  }
  logits[t] = acc;
  atomicMaxFloat(&mx[(long)d * heads + h], acc);
}

__global__ void k_edge_exp(const float* __restrict__ logits, const float* __restrict__ mx,
                           const int* __restrict__ dst,
                           float* __restrict__ ex, float* __restrict__ den,
                           int E, int heads) {
  long t = (long)blockIdx.x * blockDim.x + threadIdx.x;
  if (t >= (long)E * heads) return;
  int e = (int)(t / heads), h = (int)(t - (long)e * heads);
  long di = (long)dst[e] * heads + h;
  float v = expf(logits[t] - mx[di]);
  ex[t] = v;
  atomicAdd(&den[di], v);
}

// layer-1 scatter: one block per edge, HC threads (HC = heads*C = 128)
__global__ void k_scatter_block(const float* __restrict__ ex, const float* __restrict__ den,
                                const float* __restrict__ xl,
                                const int* __restrict__ src, const int* __restrict__ dst,
                                float* __restrict__ out, int heads, int C, int HC) {
  long e = blockIdx.x;
  int c = threadIdx.x;
  int h = c / C;
  int s = src[e], d = dst[e];
  float alpha = ex[e * heads + h] / den[(long)d * heads + h];
  atomicAdd(&out[(long)d * HC + c], alpha * xl[(long)s * HC + c]);
}

// layer-2 scatter: one thread per edge (strideX may be padded)
__global__ void k_scatter_small(const float* __restrict__ ex, const float* __restrict__ den,
                                const float* __restrict__ xl,
                                const int* __restrict__ src, const int* __restrict__ dst,
                                float* __restrict__ out, int E, int C, int strideX) {
  long e = (long)blockIdx.x * blockDim.x + threadIdx.x;
  if (e >= E) return;
  int s = src[e], d = dst[e];
  float alpha = ex[e] / den[d];
  for (int c = 0; c < C; ++c)
    atomicAdd(&out[(long)d * C + c], alpha * xl[(long)s * strideX + c]);
}

// h = elu(acc + bias), stored as f16 for the layer-2 WMMA A-matrix
__global__ void k_elu_to_f16(const float* __restrict__ acc, const float* __restrict__ bias,
                             half_t* __restrict__ out, long n, int C) {
  long i = (long)blockIdx.x * blockDim.x + threadIdx.x;
  if (i >= n) return;
  float v = acc[i] + bias[i % C];
  v = v > 0.0f ? v : (expf(v) - 1.0f);
  out[i] = (half_t)v;
}

__global__ void k_bias_out(const float* __restrict__ acc, const float* __restrict__ bias,
                           float* __restrict__ out, long n, int C) {
  long i = (long)blockIdx.x * blockDim.x + threadIdx.x;
  if (i >= n) return;
  out[i] = acc[i] + bias[i % C];
}

// ---------------- host launcher ----------------

extern "C" void kernel_launch(void* const* d_in, const int* in_sizes, int n_in,
                              void* d_out, int out_size, void* d_ws, size_t ws_size,
                              hipStream_t stream) {
  (void)n_in; (void)out_size; (void)ws_size;

  const float* x     = (const float*)d_in[0];
  const int*   eidx  = (const int*)  d_in[1];
  const float* eattr = (const float*)d_in[2];
  const float* W1l   = (const float*)d_in[3];
  const float* b1l   = (const float*)d_in[4];
  const float* W1r   = (const float*)d_in[5];
  const float* b1r   = (const float*)d_in[6];
  const float* W1e   = (const float*)d_in[7];
  const float* b1e   = (const float*)d_in[8];
  const float* att1  = (const float*)d_in[9];
  const float* bias1 = (const float*)d_in[10];
  const float* W2l   = (const float*)d_in[11];
  const float* b2l   = (const float*)d_in[12];
  const float* W2r   = (const float*)d_in[13];
  const float* b2r   = (const float*)d_in[14];
  const float* W2e   = (const float*)d_in[15];
  const float* b2e   = (const float*)d_in[16];
  const float* att2  = (const float*)d_in[17];
  const float* bias2 = (const float*)d_in[18];

  const int IN_CH = 128, HID = 64, HEADS = 2, OUT_CH = 5;
  const int HC = HEADS * HID;                 // 128
  const int NPAD2 = 16;                       // layer-2 padded N
  const int N = in_sizes[0] / IN_CH;          // 50000
  const int E = in_sizes[2];                  // 800000
  const int* src = eidx;
  const int* dst = eidx + E;

  // workspace carve-out (256B aligned)
  char* ws = (char*)d_ws;
  size_t off = 0;
  auto carve = [&](size_t bytes) -> void* {
    void* p = ws + off;
    off += (bytes + 255) & ~(size_t)255;
    return p;
  };
  half_t* x16    = (half_t*)carve((size_t)N * IN_CH * sizeof(half_t)); // reused as h16
  half_t* w1lp   = (half_t*)carve((size_t)IN_CH * HC * sizeof(half_t));  // packed
  half_t* w1rp   = (half_t*)carve((size_t)IN_CH * HC * sizeof(half_t));  // packed
  float*  xl1    = (float*) carve((size_t)N * HC * sizeof(float));
  float*  xr1    = (float*) carve((size_t)N * HC * sizeof(float));
  float*  logit1 = (float*) carve((size_t)E * HEADS * sizeof(float));
  float*  ex1    = (float*) carve((size_t)E * HEADS * sizeof(float));
  float*  mx1    = (float*) carve((size_t)N * HEADS * sizeof(float));
  float*  den1   = (float*) carve((size_t)N * HEADS * sizeof(float));
  float*  hacc   = (float*) carve((size_t)N * HC * sizeof(float));
  half_t* w2lp   = (half_t*)carve((size_t)HC * NPAD2 * sizeof(half_t));  // packed+padded
  half_t* w2rp   = (half_t*)carve((size_t)HC * NPAD2 * sizeof(half_t));  // packed+padded
  float*  b2lp   = (float*) carve(NPAD2 * sizeof(float));
  float*  b2rp   = (float*) carve(NPAD2 * sizeof(float));
  float*  xl2    = (float*) carve((size_t)N * NPAD2 * sizeof(float));
  float*  xr2    = (float*) carve((size_t)N * NPAD2 * sizeof(float));
  float*  logit2 = (float*) carve((size_t)E * sizeof(float));
  float*  ex2    = (float*) carve((size_t)E * sizeof(float));
  float*  mx2    = (float*) carve((size_t)N * sizeof(float));
  float*  den2   = (float*) carve((size_t)N * sizeof(float));
  float*  oacc   = (float*) carve((size_t)N * OUT_CH * sizeof(float));
  half_t* h16    = x16;                       // reuse: x16 dead after layer-1 GEMMs

  auto cdiv = [](long a, long b) -> int { return (int)((a + b - 1) / b); };

  // ---- init accumulators (every call: deterministic) ----
  k_fill_f32<<<cdiv((long)N * HEADS, 256), 256, 0, stream>>>(mx1, -INFINITY, (long)N * HEADS);
  k_fill_f32<<<cdiv((long)N * HEADS, 256), 256, 0, stream>>>(den1, 0.0f, (long)N * HEADS);
  k_fill_f32<<<cdiv((long)N * HC, 256), 256, 0, stream>>>(hacc, 0.0f, (long)N * HC);
  k_fill_f32<<<cdiv((long)N, 256), 256, 0, stream>>>(mx2, -INFINITY, (long)N);
  k_fill_f32<<<cdiv((long)N, 256), 256, 0, stream>>>(den2, 0.0f, (long)N);
  k_fill_f32<<<cdiv((long)N * OUT_CH, 256), 256, 0, stream>>>(oacc, 0.0f, (long)N * OUT_CH);

  // ---- layer 1 node transforms (WMMA f16 -> f32 acc) ----
  k_f32_to_f16<<<cdiv((long)N * IN_CH, 256), 256, 0, stream>>>(x, x16, (long)N * IN_CH);
  k_pack_w<<<cdiv((long)IN_CH * HC, 256), 256, 0, stream>>>(W1l, w1lp, IN_CH, HC, HC);
  k_pack_w<<<cdiv((long)IN_CH * HC, 256), 256, 0, stream>>>(W1r, w1rp, IN_CH, HC, HC);
  int waves1 = N / 16;                         // 3125 waves, each does a 16x128 strip
  k_wmma_gemm<8, 4><<<cdiv(waves1, 4), 128, 0, stream>>>(x16, w1lp, b1l, xl1, N, HC);
  k_wmma_gemm<8, 4><<<cdiv(waves1, 4), 128, 0, stream>>>(x16, w1rp, b1r, xr1, N, HC);

  // ---- layer 1 edge phase ----
  k_edge_logits_v4<<<cdiv((long)E * HEADS, 256), 256, 0, stream>>>(
      xl1, xr1, eattr, W1e, b1e, att1, src, dst, logit1, mx1, E, HEADS, HID, HC);
  k_edge_exp<<<cdiv((long)E * HEADS, 256), 256, 0, stream>>>(
      logit1, mx1, dst, ex1, den1, E, HEADS);
  k_scatter_block<<<E, HC, 0, stream>>>(ex1, den1, xl1, src, dst, hacc, HEADS, HID, HC);

  // ---- ELU + convert to f16 for layer-2 A matrix ----
  k_elu_to_f16<<<cdiv((long)N * HC, 256), 256, 0, stream>>>(hacc, bias1, h16, (long)N * HC, HC);

  // ---- layer 2 node transforms (padded N=16 tile) ----
  k_pack_w<<<cdiv((long)HC * NPAD2, 256), 256, 0, stream>>>(W2l, w2lp, HC, OUT_CH, NPAD2);
  k_pack_w<<<cdiv((long)HC * NPAD2, 256), 256, 0, stream>>>(W2r, w2rp, HC, OUT_CH, NPAD2);
  k_pad_b_f32<<<1, NPAD2, 0, stream>>>(b2l, b2lp, OUT_CH, NPAD2);
  k_pad_b_f32<<<1, NPAD2, 0, stream>>>(b2r, b2rp, OUT_CH, NPAD2);
  k_wmma_gemm<1, 4><<<cdiv(waves1, 4), 128, 0, stream>>>(h16, w2lp, b2lp, xl2, N, NPAD2);
  k_wmma_gemm<1, 4><<<cdiv(waves1, 4), 128, 0, stream>>>(h16, w2rp, b2rp, xr2, N, NPAD2);

  // ---- layer 2 edge phase (heads=1, C=5, padded node stride 16) ----
  k_edge_logits<<<cdiv((long)E, 256), 256, 0, stream>>>(
      xl2, xr2, eattr, W2e, b2e, att2, src, dst, logit2, mx2, E, 1, OUT_CH, NPAD2);
  k_edge_exp<<<cdiv((long)E, 256), 256, 0, stream>>>(logit2, mx2, dst, ex2, den2, E, 1);
  k_scatter_small<<<cdiv((long)E, 256), 256, 0, stream>>>(
      ex2, den2, xl2, src, dst, oacc, E, OUT_CH, NPAD2);

  // ---- epilogue ----
  k_bias_out<<<cdiv((long)N * OUT_CH, 256), 256, 0, stream>>>(
      oacc, bias2, (float*)d_out, (long)N * OUT_CH, OUT_CH);
}
